// Decoder_6786048327877
// MI455X (gfx1250) — matrix operand order repaired and checked by reading, not compiled, not run on hardware
//
#include <hip/hip_runtime.h>
#include <hip/hip_bf16.h>
#include <math.h>

// ---------------------------------------------------------------------------
// Types
// ---------------------------------------------------------------------------
typedef _Float16 f16;
typedef __attribute__((ext_vector_type(16))) _Float16 v16h;
typedef __attribute__((ext_vector_type(8)))  _Float16 v8h;
typedef __attribute__((ext_vector_type(4)))  _Float16 v4h;
typedef __attribute__((ext_vector_type(2)))  _Float16 v2h;
typedef __attribute__((ext_vector_type(8)))  float    v8f;
typedef __attribute__((ext_vector_type(4)))  float    v4f;

// Model dims (fixed by the reference)
#define BB   2
#define SS   2048
#define EE   2048
#define MM   (BB * SS)      // 4096 rows
#define HH   16
#define DD   128
#define FF   (4 * EE)       // 8192

// ---------------------------------------------------------------------------
// WMMA fragment loaders (CDNA5 ISA 7.12.2, wave32, f16 16x16x32).
// All loads are 16B-aligned v8h (-> ds_load_b128 / global_load_b128).
// A (16x32 MxK, row-major, stride ld): lane(m,half) = two 8-half runs.
// B (32x16 KxN) staged K-MAJOR (Bt[n][k]): lane(n,half) = two 8-half runs.
// C/D (16x16 f32): VGPR r: lanes 0..15 -> M=r, N=lane; lanes 16..31 -> M=r+8.
// ---------------------------------------------------------------------------
__device__ inline v16h frag_join(const f16* p0, const f16* p1) {
    v8h lo = *(const v8h*)p0;
    v8h hi = *(const v8h*)p1;
    return __builtin_shufflevector(lo, hi, 0, 1, 2, 3, 4, 5, 6, 7,
                                           8, 9, 10, 11, 12, 13, 14, 15);
}

__device__ inline v16h load_a_frag(const f16* base, int ld) {
    const int lane = threadIdx.x & 31;
    const f16* p = base + (lane & 15) * ld + (lane >> 4) * 8;
    return frag_join(p, p + 16);
}

__device__ inline v16h load_bt_frag(const f16* base, int ld) {
    const int lane = threadIdx.x & 31;
    const f16* p = base + (lane & 15) * ld + (lane >> 4) * 8;
    return frag_join(p, p + 16);
}

// ---------------------------------------------------------------------------
// f32 -> f16 conversion (grid-stride, 4-wide)
// ---------------------------------------------------------------------------
__global__ __launch_bounds__(256) void f32_to_f16_kernel(
    const float* __restrict__ in, f16* __restrict__ out, long n4) {
    long i      = (long)blockIdx.x * blockDim.x + threadIdx.x;
    long stride = (long)gridDim.x * blockDim.x;
    for (; i < n4; i += stride) {
        const v4f v = *(const v4f*)&in[i * 4];
        v4h o; o[0] = (f16)v[0]; o[1] = (f16)v[1]; o[2] = (f16)v[2]; o[3] = (f16)v[3];
        *(v4h*)&out[i * 4] = o;
    }
}

// ---------------------------------------------------------------------------
// Tiled WMMA GEMM: C[M,N] = A[M,K] @ B[K,N] + bias (+ residual / GELU)
//   block tile 128x128, K-round 64 (2 wmma k-steps), 256 thr = 8 waves,
//   wave tile 32x64 -> 16 WMMAs per round.  Software-pipelined staging:
//   next tile's global loads are issued before the current tile's WMMAs.
//   f16 outputs are transposed through LDS for coalesced b128 stores.
//   mode 0: store f16; 1: store f16 + exact GELU; 2: store f32 + residual.
// ---------------------------------------------------------------------------
#define GBM 128
#define GBN 128
#define GBK 64
#define LDA 72    // 64 + 8 halves
#define LDT 72    // Bt rows: 64 k + 8 pad
#define LDC 136   // epilogue C tile: 128 + 8 pad

__global__ __launch_bounds__(256) void gemm_wmma_kernel(
    const f16* __restrict__ A, const f16* __restrict__ B,
    const float* __restrict__ bias, const float* __restrict__ res,
    f16* __restrict__ out16, float* __restrict__ out32,
    int M, int N, int K, int mode) {
    __shared__ __align__(16) union {
        struct { f16 a[GBM * LDA]; f16 b[GBN * LDT]; } s;  // 36864 B
        f16 c[GBM * LDC];                                  // 34816 B
    } sm;
    f16* const As = sm.s.a;
    f16* const Bt = sm.s.b;

    const int tid  = threadIdx.x;
    const int lane = tid & 31;
    const int wid  = tid >> 5;
    const int wrow = (wid & 3) * 32;
    const int wcol = (wid >> 2) * 64;
    const int bm   = blockIdx.y * GBM;
    const int bn   = blockIdx.x * GBN;

    v8f acc[2][4] = {};
    v8h stA[4];      // staged A chunks (row-major)
    v8h stB[2][2];   // staged B chunks (2 k-rows x 8 n-cols each)

    auto load_stage = [&](int k0) {
#pragma unroll
        for (int c = 0; c < 4; ++c) {
            const int e = (tid + c * 256) * 8;
            stA[c] = *(const v8h*)&A[(size_t)(bm + (e >> 6)) * K + k0 + (e & 63)];
        }
#pragma unroll
        for (int c = 0; c < 2; ++c) {
            const int pc = tid + c * 256;          // 0..511
            const int k  = (pc >> 4) * 2;          // even 0..62
            const int nc = (pc & 15) * 8;          // 0..120
            stB[c][0] = *(const v8h*)&B[(size_t)(k0 + k) * N + bn + nc];
            stB[c][1] = *(const v8h*)&B[(size_t)(k0 + k + 1) * N + bn + nc];
        }
    };
    auto store_stage = [&]() {
#pragma unroll
        for (int c = 0; c < 4; ++c) {
            const int e = (tid + c * 256) * 8;
            *(v8h*)&As[(e >> 6) * LDA + (e & 63)] = stA[c];
        }
#pragma unroll
        for (int c = 0; c < 2; ++c) {
            const int pc = tid + c * 256;
            const int k  = (pc >> 4) * 2;
            const int nc = (pc & 15) * 8;
#pragma unroll
            for (int j = 0; j < 8; ++j) {
                v2h pr = { stB[c][0][j], stB[c][1][j] };
                *(v2h*)&Bt[(nc + j) * LDT + k] = pr;
            }
        }
    };

    load_stage(0);
    store_stage();
    __syncthreads();

    for (int k0 = 0; k0 < K; k0 += GBK) {
        const bool more = (k0 + GBK < K);
        if (more) load_stage(k0 + GBK);  // issue globals before compute

#pragma unroll
        for (int kc = 0; kc < 2; ++kc) {
            const v16h af0 = load_a_frag(&As[(wrow + 0)  * LDA + kc * 32], LDA);
            const v16h af1 = load_a_frag(&As[(wrow + 16) * LDA + kc * 32], LDA);
            v16h bf[4];
#pragma unroll
            for (int ni = 0; ni < 4; ++ni)
                bf[ni] = load_bt_frag(&Bt[(wcol + ni * 16) * LDT + kc * 32], LDT);
#pragma unroll
            for (int ni = 0; ni < 4; ++ni) {
                acc[0][ni] = __builtin_amdgcn_wmma_f32_16x16x32_f16(
                    false, af0, false, bf[ni], (short)0, acc[0][ni], false, false);
                acc[1][ni] = __builtin_amdgcn_wmma_f32_16x16x32_f16(
                    false, af1, false, bf[ni], (short)0, acc[1][ni], false, false);
            }
        }
        if (more) {
            __syncthreads();
            store_stage();
            __syncthreads();
        }
    }

    // Epilogue.
    const int n    = lane & 15;
    const int half = lane >> 4;
    if (mode != 2) {
        // f16 output: transpose through LDS for coalesced b128 global stores.
        __syncthreads();
#pragma unroll
        for (int mi = 0; mi < 2; ++mi) {
#pragma unroll
            for (int ni = 0; ni < 4; ++ni) {
#pragma unroll
                for (int r = 0; r < 8; ++r) {
                    const int lrow = wrow + mi * 16 + r + half * 8;
                    const int lcol = wcol + ni * 16 + n;
                    float v = acc[mi][ni][r] + bias[bn + lcol];
                    if (mode == 1) v = 0.5f * v * (1.0f + erff(v * 0.70710678118654752f));
                    sm.c[lrow * LDC + lcol] = (f16)v;
                }
            }
        }
        __syncthreads();
#pragma unroll
        for (int c = 0; c < 8; ++c) {
            const int e = (tid + c * 256) * 8;
            const int r = e >> 7, col = e & 127;
            *(v8h*)&out16[(size_t)(bm + r) * N + bn + col] = *(const v8h*)&sm.c[r * LDC + col];
        }
    } else {
        // f32 output + residual: direct stores.
#pragma unroll
        for (int mi = 0; mi < 2; ++mi) {
#pragma unroll
            for (int ni = 0; ni < 4; ++ni) {
#pragma unroll
                for (int r = 0; r < 8; ++r) {
                    const int grow = bm + wrow + mi * 16 + r + half * 8;
                    const int gcol = bn + wcol + ni * 16 + n;
                    const size_t idx = (size_t)grow * N + gcol;
                    out32[idx] = acc[mi][ni][r] + bias[gcol] + res[idx];
                }
            }
        }
    }
}

// ---------------------------------------------------------------------------
// Causal flash attention, WMMA everywhere.
//   grid (S/64, H, B); 128 thr = 4 waves; wave handles 16 q rows.
//   K tile row-major in LDS (QK^T B-frags contiguous in d);
//   V tile transposed in LDS (PV B-frags contiguous in t);
//   P round-trips through LDS into A-fragment layout; ctx output goes
//   through LDS (reusing the K tile) for coalesced b128 stores.
// ---------------------------------------------------------------------------
#define ALD  136   // K tile: 128 d + 8 pad
#define LDVT 72    // V^T tile rows: 64 t + 8 pad
#define PLD  72    // P tile rows: 64 t + 8 pad

__global__ __launch_bounds__(128) void attn_wmma_kernel(
    const f16* __restrict__ Qh, const f16* __restrict__ Kh,
    const f16* __restrict__ Vh, f16* __restrict__ ctx) {
    __shared__ __align__(16) f16 Kt[64 * ALD];    // [t][d]; reused for ctx out
    __shared__ __align__(16) f16 Vt[128 * LDVT];  // [d][t]  (transposed)
    __shared__ __align__(16) f16 Pl[4][16 * PLD]; // per-wave [m][t]

    const int tid   = threadIdx.x;
    const int lane  = tid & 31;
    const int wid   = tid >> 5;
    const int n     = lane & 15;
    const int half  = lane >> 4;
    const int qbase = blockIdx.x * 64;
    const int h     = blockIdx.y;
    const int b     = blockIdx.z;
    const int qrow0 = qbase + wid * 16;
    const size_t headcol = (size_t)h * DD;

    // Q fragments (16 rows x 128 d = 4 A-frags), straight from global.
    v16h aq[4];
#pragma unroll
    for (int kt = 0; kt < 4; ++kt)
        aq[kt] = load_a_frag(&Qh[(size_t)(b * SS + qrow0) * EE + headcol + kt * 32], EE);

    v8f cacc[8] = {};
    float mrun[8], lrun[8];
#pragma unroll
    for (int r = 0; r < 8; ++r) { mrun[r] = -3.0e38f; lrun[r] = 0.0f; }

    const float scale = 0.08838834764831845f;  // 1/sqrt(128)

    for (int t0 = 0; t0 <= qbase; t0 += 64) {
        // Stage K/V: issue all global loads first, then all LDS stores.
        v8h kreg[8];
        v8h vreg[4][2];
#pragma unroll
        for (int c = 0; c < 8; ++c) {
            const int e = (tid + c * 128) * 8;
            kreg[c] = *(const v8h*)&Kh[(size_t)(b * SS + t0 + (e >> 7)) * EE + headcol + (e & 127)];
        }
#pragma unroll
        for (int c = 0; c < 4; ++c) {
            const int pc = tid + c * 128;          // 0..511
            const int t  = (pc >> 4) * 2;          // even 0..62
            const int dc = (pc & 15) * 8;
            const size_t g = (size_t)(b * SS + t0 + t) * EE + headcol + dc;
            vreg[c][0] = *(const v8h*)&Vh[g];
            vreg[c][1] = *(const v8h*)&Vh[g + EE];
        }
#pragma unroll
        for (int c = 0; c < 8; ++c) {
            const int e = (tid + c * 128) * 8;
            *(v8h*)&Kt[(e >> 7) * ALD + (e & 127)] = kreg[c];
        }
#pragma unroll
        for (int c = 0; c < 4; ++c) {
            const int pc = tid + c * 128;
            const int t  = (pc >> 4) * 2;
            const int dc = (pc & 15) * 8;
#pragma unroll
            for (int j = 0; j < 8; ++j) {
                v2h pr = { vreg[c][0][j], vreg[c][1][j] };
                *(v2h*)&Vt[(dc + j) * LDVT + t] = pr;
            }
        }
        __syncthreads();

        // Scores S = Q K^T: B-frag (kd,n) = Kt[ni*16+n][kt*32+kd] -> contiguous.
        v8f sacc[4] = {};
#pragma unroll
        for (int ni = 0; ni < 4; ++ni) {
#pragma unroll
            for (int kt = 0; kt < 4; ++kt) {
                const v16h bfr = load_bt_frag(&Kt[(ni * 16) * ALD + kt * 32], ALD);
                sacc[ni] = __builtin_amdgcn_wmma_f32_16x16x32_f16(
                    false, aq[kt], false, bfr, (short)0, sacc[ni], false, false);
            }
        }

        // Causal mask + scale + online softmax per row; write P (f16) to LDS.
#pragma unroll
        for (int r = 0; r < 8; ++r) {
            const int srow = qrow0 + r + half * 8;
            float pv[4];
            float rmax = -3.0e38f;
#pragma unroll
            for (int ni = 0; ni < 4; ++ni) {
                const int tcol = t0 + ni * 16 + n;
                const float v = (tcol <= srow) ? sacc[ni][r] * scale : -3.0e38f;
                pv[ni] = v;
                rmax   = fmaxf(rmax, v);
            }
#pragma unroll
            for (int o = 1; o < 16; o <<= 1)
                rmax = fmaxf(rmax, __shfl_xor(rmax, o, 32));
            const float mnew = fmaxf(mrun[r], rmax);
            const float corr = __expf(mrun[r] - mnew);
            float rsum = 0.0f;
#pragma unroll
            for (int ni = 0; ni < 4; ++ni) {
                const float p = __expf(pv[ni] - mnew);
                rsum += p;
                Pl[wid][(r + half * 8) * PLD + ni * 16 + n] = (f16)p;
            }
#pragma unroll
            for (int o = 1; o < 16; o <<= 1)
                rsum += __shfl_xor(rsum, o, 32);
            lrun[r] = lrun[r] * corr + rsum;
            mrun[r] = mnew;
#pragma unroll
            for (int dni = 0; dni < 8; ++dni) cacc[dni][r] *= corr;
        }
        __syncthreads();

        // ctx += P @ V: A-frags from Pl, B-frags contiguous rows of Vt.
#pragma unroll
        for (int kt2 = 0; kt2 < 2; ++kt2) {
            const v16h ap = load_a_frag(&Pl[wid][kt2 * 32], PLD);
#pragma unroll
            for (int dni = 0; dni < 8; ++dni) {
                const v16h bfr = load_bt_frag(&Vt[(dni * 16) * LDVT + kt2 * 32], LDVT);
                cacc[dni] = __builtin_amdgcn_wmma_f32_16x16x32_f16(
                    false, ap, false, bfr, (short)0, cacc[dni], false, false);
            }
        }
        __syncthreads();
    }

    // Normalize, transpose through LDS (reusing Kt), store coalesced b128.
#pragma unroll
    for (int dni = 0; dni < 8; ++dni) {
#pragma unroll
        for (int r = 0; r < 8; ++r) {
            const float v = cacc[dni][r] / lrun[r];
            Kt[(wid * 16 + r + half * 8) * ALD + dni * 16 + n] = (f16)v;
        }
    }
    __syncthreads();
#pragma unroll
    for (int c = 0; c < 8; ++c) {
        const int e = (tid + c * 128) * 8;
        const int r = e >> 7, col = e & 127;
        *(v8h*)&ctx[(size_t)(b * SS + qbase + r) * EE + headcol + col] =
            *(const v8h*)&Kt[r * ALD + col];
    }
}

// ---------------------------------------------------------------------------
// LayerNorm over last dim (N = 2048), one block (256 thr) per row.
// ---------------------------------------------------------------------------
__global__ __launch_bounds__(256) void layernorm_kernel(
    const float* __restrict__ x, const float* __restrict__ g,
    const float* __restrict__ be, float* __restrict__ y32,
    f16* __restrict__ y16, int N) {
    __shared__ float red[8];
    const int row  = blockIdx.x;
    const int tid  = threadIdx.x;
    const int lane = tid & 31;
    const int wid  = tid >> 5;
    const float* xr = x + (size_t)row * N;

    float vals[8];
    float s = 0.0f;
#pragma unroll
    for (int i = 0; i < 8; ++i) { vals[i] = xr[tid + i * 256]; s += vals[i]; }
#pragma unroll
    for (int o = 1; o < 32; o <<= 1) s += __shfl_xor(s, o, 32);
    if (lane == 0) red[wid] = s;
    __syncthreads();
    float mean = 0.0f;
#pragma unroll
    for (int i = 0; i < 8; ++i) mean += red[i];
    mean /= (float)N;

    float vs = 0.0f;
#pragma unroll
    for (int i = 0; i < 8; ++i) { const float d = vals[i] - mean; vs += d * d; }
#pragma unroll
    for (int o = 1; o < 32; o <<= 1) vs += __shfl_xor(vs, o, 32);
    __syncthreads();
    if (lane == 0) red[wid] = vs;
    __syncthreads();
    float var = 0.0f;
#pragma unroll
    for (int i = 0; i < 8; ++i) var += red[i];
    const float inv = rsqrtf(var / (float)N + 1e-5f);

#pragma unroll
    for (int i = 0; i < 8; ++i) {
        const int c   = tid + i * 256;
        const float o = (vals[i] - mean) * inv * g[c] + be[c];
        if (y32) y32[(size_t)row * N + c] = o;
        if (y16) y16[(size_t)row * N + c] = (f16)o;
    }
}

// ---------------------------------------------------------------------------
// Host orchestration
// ---------------------------------------------------------------------------
extern "C" void kernel_launch(void* const* d_in, const int* in_sizes, int n_in,
                              void* d_out, int out_size, void* d_ws, size_t ws_size,
                              hipStream_t stream) {
    (void)in_sizes; (void)n_in; (void)out_size; (void)ws_size;

    const float* emb = (const float*)d_in[0];
    const float* Wq  = (const float*)d_in[1];
    const float* bq  = (const float*)d_in[2];
    const float* Wk  = (const float*)d_in[3];
    const float* bk  = (const float*)d_in[4];
    const float* Wv  = (const float*)d_in[5];
    const float* bv  = (const float*)d_in[6];
    const float* Wfc = (const float*)d_in[7];
    const float* bfc = (const float*)d_in[8];
    const float* g1  = (const float*)d_in[9];
    const float* be1 = (const float*)d_in[10];
    const float* W1  = (const float*)d_in[11];
    const float* b1  = (const float*)d_in[12];
    const float* W2  = (const float*)d_in[13];
    const float* b2  = (const float*)d_in[14];
    const float* g2  = (const float*)d_in[15];
    const float* be2 = (const float*)d_in[16];

    char*  ws  = (char*)d_ws;
    size_t off = 0;
    auto alloc = [&](size_t bytes) -> char* {
        char* p = ws + off;
        off += (bytes + 255) & ~(size_t)255;
        return p;
    };
    f16*   xh    = (f16*)alloc((size_t)MM * EE * 2);
    f16*   wq16  = (f16*)alloc((size_t)EE * EE * 2);
    f16*   wk16  = (f16*)alloc((size_t)EE * EE * 2);
    f16*   wv16  = (f16*)alloc((size_t)EE * EE * 2);
    f16*   wfc16 = (f16*)alloc((size_t)EE * EE * 2);
    f16*   w116  = (f16*)alloc((size_t)EE * FF * 2);
    f16*   w216  = (f16*)alloc((size_t)FF * EE * 2);
    f16*   qh    = (f16*)alloc((size_t)MM * EE * 2);
    f16*   kh    = (f16*)alloc((size_t)MM * EE * 2);
    f16*   vh    = (f16*)alloc((size_t)MM * EE * 2);
    float* x1    = (float*)alloc((size_t)MM * EE * 4);
    float* y1    = (float*)alloc((size_t)MM * EE * 4);
    f16*   hh    = (f16*)alloc((size_t)MM * FF * 2);
    f16*   ctxh  = xh;  // embeddings-f16 dead after QKV GEMMs
    f16*   y1h   = qh;  // Q dead after attention
    float* z     = x1;  // x1 dead after LN1

    auto cvt = [&](const float* src, f16* dst, long nelem) {
        const long n4 = nelem / 4;
        const int blocks = (int)((n4 + 256L * 4 - 1) / (256L * 4));
        f32_to_f16_kernel<<<blocks, 256, 0, stream>>>(src, dst, n4);
    };
    cvt(emb, xh,    (long)MM * EE);
    cvt(Wq,  wq16,  (long)EE * EE);
    cvt(Wk,  wk16,  (long)EE * EE);
    cvt(Wv,  wv16,  (long)EE * EE);
    cvt(Wfc, wfc16, (long)EE * EE);
    cvt(W1,  w116,  (long)EE * FF);
    cvt(W2,  w216,  (long)FF * EE);

    const dim3 blk(256);
    const dim3 gEE(EE / GBN, MM / GBM);
    const dim3 gFF(FF / GBN, MM / GBM);

    // QKV projections (f16 out).
    gemm_wmma_kernel<<<gEE, blk, 0, stream>>>(xh, wq16, bq, nullptr, qh, nullptr, MM, EE, EE, 0);
    gemm_wmma_kernel<<<gEE, blk, 0, stream>>>(xh, wk16, bk, nullptr, kh, nullptr, MM, EE, EE, 0);
    gemm_wmma_kernel<<<gEE, blk, 0, stream>>>(xh, wv16, bv, nullptr, vh, nullptr, MM, EE, EE, 0);

    // Causal flash attention -> ctx (f16).
    attn_wmma_kernel<<<dim3(SS / 64, HH, BB), dim3(128), 0, stream>>>(qh, kh, vh, ctxh);

    // Out-projection + residual (f32), then LN1 (f32 + f16 copies).
    gemm_wmma_kernel<<<gEE, blk, 0, stream>>>(ctxh, wfc16, bfc, emb, nullptr, x1, MM, EE, EE, 2);
    layernorm_kernel<<<MM, 256, 0, stream>>>(x1, g1, be1, y1, y1h, EE);

    // FFN: up-proj with exact GELU (f16), down-proj + residual (f32), LN2 -> out.
    gemm_wmma_kernel<<<gFF, blk, 0, stream>>>(y1h, w116, b1, nullptr, hh, nullptr, MM, FF, EE, 1);
    gemm_wmma_kernel<<<gEE, blk, 0, stream>>>(hh, w216, b2, y1, nullptr, z, MM, EE, FF, 2);
    layernorm_kernel<<<MM, 256, 0, stream>>>(z, g2, be2, (float*)d_out, nullptr, EE);
}
